// CNNadgnn_cnn1ch_34479997452439
// MI455X (gfx1250) — matrix operand order repaired and checked by reading, not compiled, not run on hardware
//
#include <hip/hip_runtime.h>

#define N_NODES 100000
#define N_EDGES 800000
#define N_GRAPHS 64
#define HW 128

typedef __attribute__((ext_vector_type(16))) _Float16 v16h;
typedef __attribute__((ext_vector_type(8)))  float    v8f;

// ---------------- conv1 effective weights: sum over the 6 repeated in-channels
__global__ __launch_bounds__(144) void prep_weff_k(const float* __restrict__ w1,
                                                   float* __restrict__ weff) {
  int t = threadIdx.x;
  if (t < 144) {
    int o = t / 9, k = t % 9;
    float s = 0.f;
#pragma unroll
    for (int i = 0; i < 6; ++i) s += w1[o*54 + i*9 + k];
    weff[t] = s;
  }
}

// ---------------- fully fused 4-layer conv stack, one 8x8 output tile per block
__global__ __launch_bounds__(256) void conv_fused_k(
    const float* __restrict__ x, const float* __restrict__ weff,
    const float* __restrict__ b1, const float* __restrict__ w2, const float* __restrict__ b2,
    const float* __restrict__ w3, const float* __restrict__ b3,
    const float* __restrict__ w4, const float* __restrict__ b4,
    float* __restrict__ out)
{
  __shared__ float xs[16][16];
  __shared__ float c1s[14*14*16];   // reused for c3 (10*10*16)
  __shared__ float c2s[12*12*32];
  __shared__ float wbuf[4608];      // transposed w2, then transposed w3
  __shared__ float weffs[144];
  __shared__ float w4s[144];

  int tid = threadIdx.x;
  int g   = blockIdx.z;
  int ty0 = blockIdx.y * 8, tx0 = blockIdx.x * 8;
  const float* xg = x + g * (HW*HW);

  { int ly = tid >> 4, lx = tid & 15;
    int gy = ty0 - 4 + ly, gx = tx0 - 4 + lx;
    xs[ly][lx] = (gy >= 0 && gy < HW && gx >= 0 && gx < HW) ? xg[gy*HW + gx] : 0.f; }
  for (int i = tid; i < 144; i += 256) { weffs[i] = weff[i]; w4s[i] = w4[i]; }
  // w2 (32,16,3,3) -> wbuf[(ic*9+k)*32 + oc]  (conflict-free oc reads)
  for (int i = tid; i < 4608; i += 256) { int o = i/144, rem = i%144; wbuf[rem*32 + o] = w2[i]; }
  __syncthreads();

  // c1: 14x14x16, global y in [ty0-3, ty0+10]; SAME padding => zero outside image
  for (int idx = tid; idx < 14*14*16; idx += 256) {
    int ch = idx & 15, p = idx >> 4;
    int ry = p / 14, rx = p % 14;
    int gy = ty0 - 3 + ry, gx = tx0 - 3 + rx;
    float s = 0.f;
    if (gy >= 0 && gy < HW && gx >= 0 && gx < HW) {
      s = b1[ch];
#pragma unroll
      for (int ky = 0; ky < 3; ++ky)
#pragma unroll
        for (int kx = 0; kx < 3; ++kx)
          s += xs[ry+ky][rx+kx] * weffs[ch*9 + ky*3 + kx];
      s = fmaxf(s, 0.f);
    }
    c1s[p*16 + ch] = s;
  }
  __syncthreads();

  // c2: 12x12x32
  for (int idx = tid; idx < 12*12*32; idx += 256) {
    int oc = idx & 31, p = idx >> 5;
    int ry = p / 12, rx = p % 12;
    int gy = ty0 - 2 + ry, gx = tx0 - 2 + rx;
    float s = 0.f;
    if (gy >= 0 && gy < HW && gx >= 0 && gx < HW) {
      s = b2[oc];
#pragma unroll
      for (int ic = 0; ic < 16; ++ic)
#pragma unroll
        for (int ky = 0; ky < 3; ++ky)
#pragma unroll
          for (int kx = 0; kx < 3; ++kx)
            s += c1s[((ry+ky)*14 + rx+kx)*16 + ic] * wbuf[(ic*9 + ky*3 + kx)*32 + oc];
      s = fmaxf(s, 0.f);
    }
    c2s[p*32 + oc] = s;
  }
  __syncthreads();
  // w3 (16,32,3,3) -> wbuf[(ic*9+k)*16 + oc]
  for (int i = tid; i < 4608; i += 256) { int o = i/288, rem = i%288; wbuf[rem*16 + o] = w3[i]; }
  __syncthreads();

  // c3: 10x10x16 into c1s (c1 no longer needed)
  for (int idx = tid; idx < 10*10*16; idx += 256) {
    int oc = idx & 15, p = idx >> 4;
    int ry = p / 10, rx = p % 10;
    int gy = ty0 - 1 + ry, gx = tx0 - 1 + rx;
    float s = 0.f;
    if (gy >= 0 && gy < HW && gx >= 0 && gx < HW) {
      s = b3[oc];
#pragma unroll
      for (int ic = 0; ic < 32; ++ic)
#pragma unroll
        for (int ky = 0; ky < 3; ++ky)
#pragma unroll
          for (int kx = 0; kx < 3; ++kx)
            s += c2s[((ry+ky)*12 + rx+kx)*32 + ic] * wbuf[(ic*9 + ky*3 + kx)*16 + oc];
      s = fmaxf(s, 0.f);
    }
    c1s[p*16 + oc] = s;
  }
  __syncthreads();

  // c4: 8x8 final output (no relu)
  if (tid < 64) {
    int ry = tid >> 3, rx = tid & 7;
    float s = b4[0];
#pragma unroll
    for (int ic = 0; ic < 16; ++ic)
#pragma unroll
      for (int ky = 0; ky < 3; ++ky)
#pragma unroll
        for (int kx = 0; kx < 3; ++kx)
          s += c1s[((ry+ky)*10 + rx+kx)*16 + ic] * w4s[ic*9 + ky*3 + kx];
    out[g*(HW*HW) + (ty0+ry)*HW + (tx0+rx)] = s;
  }
}

// ---------------- stacked [wl; wr] weights -> f16, K padded, N padded to 128
__global__ __launch_bounds__(256) void prep_w_k(const float* __restrict__ wl,
                                                const float* __restrict__ wr,
                                                _Float16* __restrict__ W,
                                                int din, int dout, int kpad)
{
  int t = blockIdx.x * 256 + threadIdx.x;
  if (t >= kpad * 128) return;
  int k = t >> 7, j = t & 127;
  float v = 0.f;
  if (j < dout) {
    if (k < din)            v = wl[k*dout + j];
    else if (k < 2*din)     v = wr[(k-din)*dout + j];
  }
  W[t] = (_Float16)v;
}

// ---------------- build h0 = [x_nodes(5), pos(2), gathered cnn feature]
__global__ __launch_bounds__(256) void build_h0_k(const float* __restrict__ xn,
                                                  const float* __restrict__ pos,
                                                  const int* __restrict__ batch,
                                                  const float* __restrict__ x1,
                                                  float* __restrict__ h)
{
  int n = blockIdx.x * 256 + threadIdx.x;
  if (n >= N_NODES) return;
  float px = pos[n*2+0], py = pos[n*2+1];
  int ix = (int)rintf(px * (127.0f / 20.0f));   // round-half-even matches jnp.round
  int iy = (int)rintf(py * (127.0f / 20.0f));
  ix = min(max(ix, 0), 127); iy = min(max(iy, 0), 127);
  int b = batch[n];
  float v = x1[b*(HW*HW) + iy*HW + ix];
  float* hr = h + (long)n * 128;
  hr[0] = xn[n*5+0]; hr[1] = xn[n*5+1]; hr[2] = xn[n*5+2];
  hr[3] = xn[n*5+3]; hr[4] = xn[n*5+4];
  hr[5] = px; hr[6] = py; hr[7] = v;
}

__global__ __launch_bounds__(256) void fill_zero_k(float* __restrict__ p, int n) {
  int t = blockIdx.x * 256 + threadIdx.x;
  if (t < n) p[t] = 0.f;
}

__global__ __launch_bounds__(256) void deg_count_k(const int* __restrict__ dst,
                                                   float* __restrict__ deg) {
  int t = blockIdx.x * 256 + threadIdx.x;
  if (t < N_EDGES) atomicAdd(&deg[dst[t]], 1.0f);
}

// segment_sum over edges: one thread per (edge, feature); L2-resident atomics
__global__ __launch_bounds__(256) void scatter_k(const int* __restrict__ src,
                                                 const int* __restrict__ dst,
                                                 const float* __restrict__ h,
                                                 float* __restrict__ agg,
                                                 int din, int total)
{
  int t = blockIdx.x * 256 + threadIdx.x;
  if (t >= total) return;
  int e = t / din, f = t - e * din;
  atomicAdd(&agg[(long)dst[e]*128 + f], h[(long)src[e]*128 + f]);
}

// ---------------- fused SAGE GEMM: out = relu?([mean|x] @ [wl;wr] + b)
// wave32 per 16-row node tile. A fragments built from branch-free pointer
// selects + vector b128 loads (runs of 8 consecutive k per half-fragment),
// kept in registers across all N-tiles. All shape params compile-time so the
// dout/ldo guards fold away for the full-width layers.
template<int DIN, int KTILES, int RELU, int NTILES, int DOUT, int LDO>
__global__ __launch_bounds__(256) void sage_wmma_k(
    const float* __restrict__ agg, const float* __restrict__ deg,
    const float* __restrict__ hin, const _Float16* __restrict__ W,
    const float* __restrict__ bias, float* __restrict__ out)
{
  int wave = threadIdx.x >> 5;
  int lane = threadIdx.x & 31;
  int tile = blockIdx.x * 8 + wave;
  if (tile >= N_NODES / 16) return;          // wave-uniform; exiting waves never hit WMMA
  int m = lane & 15;
  int g = lane >> 4;
  int node = tile * 16 + m;
  float inv = 1.0f / fmaxf(deg[node], 1.0f);
  const float* aggr = agg + (long)node * 128;
  const float* hr   = hin + (long)node * 128;

  // A: 16x32 f16 per K-slice. ISA layout: halves [0..7] = k0..k0+7,
  // halves [8..15] = k0+16..k0+23, with k0 = kt*32 + 8g. Each 8-run lies
  // entirely in one region (mean / x / zero-pad) for both DIN=8 and DIN=128,
  // so pick base pointer + scale branch-free and load b128 pairs.
  v16h A[KTILES];
#pragma unroll
  for (int kt = 0; kt < KTILES; ++kt) {
    v16h a;
#pragma unroll
    for (int half = 0; half < 2; ++half) {
      int k0 = kt * 32 + half * 16 + 8 * g;
      const float* base;
      float scale;
      if (k0 < DIN)            { base = aggr + k0;        scale = inv; }   // mean
      else if (k0 < 2 * DIN)   { base = hr + (k0 - DIN);  scale = 1.f; }   // x
      else                     { base = aggr;             scale = 0.f; }   // K-pad (safe addr)
      float4 lo = *(const float4*)(base);
      float4 hi = *(const float4*)(base + 4);
      a[half*8 + 0] = (_Float16)(lo.x * scale);
      a[half*8 + 1] = (_Float16)(lo.y * scale);
      a[half*8 + 2] = (_Float16)(lo.z * scale);
      a[half*8 + 3] = (_Float16)(lo.w * scale);
      a[half*8 + 4] = (_Float16)(hi.x * scale);
      a[half*8 + 5] = (_Float16)(hi.y * scale);
      a[half*8 + 6] = (_Float16)(hi.z * scale);
      a[half*8 + 7] = (_Float16)(hi.w * scale);
    }
    A[kt] = a;
  }

  for (int nt = 0; nt < NTILES; ++nt) {
    int jn = nt * 16 + m;                    // D layout: N = lane%16
    float bj = (jn < DOUT) ? bias[jn] : 0.f; // folds to unconditional when DOUT==128
    v8f acc = {bj, bj, bj, bj, bj, bj, bj, bj};
    const _Float16* Wb = W + nt * 16;
#pragma unroll
    for (int kt = 0; kt < KTILES; ++kt) {
      // B layout: lane holds N = lane%16, halves [0..7] = k0..k0+7, [8..15] = k0+16..k0+23, k0 = kt*32 + 8g
      v16h b;
#pragma unroll
      for (int s = 0; s < 16; ++s) {
        int kk = kt * 32 + 8 * g + (s & 7) + ((s >> 3) << 4);
        b[s] = Wb[(long)kk * 128 + m];
      }
      acc = __builtin_amdgcn_wmma_f32_16x16x32_f16(false, A[kt], false, b,
                                                   (short)0, acc, false, false);
    }
#pragma unroll
    for (int r = 0; r < 8; ++r) {            // D: M = r + 8*(lane/16)
      if (jn < DOUT) {                       // folds away when DOUT==128
        int row = tile * 16 + r + 8 * g;
        float v = acc[r];
        if (RELU) v = fmaxf(v, 0.f);
        out[(long)row * LDO + jn] = v;
      }
    }
  }
}

extern "C" void kernel_launch(void* const* d_in, const int* in_sizes, int n_in,
                              void* d_out, int out_size, void* d_ws, size_t ws_size,
                              hipStream_t stream)
{
  (void)in_sizes; (void)n_in; (void)out_size; (void)ws_size;
  const float* x     = (const float*)d_in[0];
  const float* xn    = (const float*)d_in[1];
  const float* pos   = (const float*)d_in[2];
  const int*   ei    = (const int*)d_in[3];
  const int*   batch = (const int*)d_in[4];
  const float* w1 = (const float*)d_in[5];  const float* b1 = (const float*)d_in[6];
  const float* w2 = (const float*)d_in[7];  const float* b2 = (const float*)d_in[8];
  const float* w3 = (const float*)d_in[9];  const float* b3 = (const float*)d_in[10];
  const float* w4 = (const float*)d_in[11]; const float* b4 = (const float*)d_in[12];
  const float* wl[4] = {(const float*)d_in[13], (const float*)d_in[16],
                        (const float*)d_in[19], (const float*)d_in[22]};
  const float* wr[4] = {(const float*)d_in[14], (const float*)d_in[17],
                        (const float*)d_in[20], (const float*)d_in[23]};
  const float* bs[4] = {(const float*)d_in[15], (const float*)d_in[18],
                        (const float*)d_in[21], (const float*)d_in[24]};

  float* x1   = (float*)d_out;                       // (64,1,128,128)
  float* gout = (float*)d_out + N_GRAPHS * HW * HW;  // (100000,1)

  char* ws = (char*)d_ws;
  size_t off = 0;
  auto take = [&](size_t bytes) {
    char* p = ws + off;
    off = (off + bytes + 255) & ~(size_t)255;
    return p;
  };
  float*    hA   = (float*)take((size_t)N_NODES * 128 * 4);
  float*    hB   = (float*)take((size_t)N_NODES * 128 * 4);
  float*    agg  = (float*)take((size_t)N_NODES * 128 * 4);
  float*    deg  = (float*)take((size_t)N_NODES * 4);
  _Float16* W0   = (_Float16*)take(32  * 128 * 2);
  _Float16* W1   = (_Float16*)take(256 * 128 * 2);
  _Float16* W2   = (_Float16*)take(256 * 128 * 2);
  _Float16* W3   = (_Float16*)take(256 * 128 * 2);
  float*    weff = (float*)take(144 * 4);

  const int* srcp = ei;
  const int* dstp = ei + N_EDGES;

  // CNN
  prep_weff_k<<<1, 144, 0, stream>>>(w1, weff);
  conv_fused_k<<<dim3(16, 16, 64), 256, 0, stream>>>(x, weff, b1, w2, b2, w3, b3, w4, b4, x1);

  // f16 weight staging
  prep_w_k<<<(32*128 + 255)/256, 256, 0, stream>>>(wl[0], wr[0], W0,   8, 128,  32);
  prep_w_k<<<(256*128 + 255)/256, 256, 0, stream>>>(wl[1], wr[1], W1, 128, 128, 256);
  prep_w_k<<<(256*128 + 255)/256, 256, 0, stream>>>(wl[2], wr[2], W2, 128, 128, 256);
  prep_w_k<<<(256*128 + 255)/256, 256, 0, stream>>>(wl[3], wr[3], W3, 128,   1, 256);

  // graph features + degrees
  build_h0_k<<<(N_NODES + 255)/256, 256, 0, stream>>>(xn, pos, batch, x1, hA);
  fill_zero_k<<<(N_NODES + 255)/256, 256, 0, stream>>>(deg, N_NODES);
  deg_count_k<<<(N_EDGES + 255)/256, 256, 0, stream>>>(dstp, deg);

  const int gemm_blocks = (N_NODES/16 + 7) / 8;        // 782 blocks x 8 waves
  const int aggN        = N_NODES * 128;
  const int zeroBlocks  = (aggN + 255) / 256;
  const int scat8       = (N_EDGES * 8   + 255) / 256;
  const int scat128     = (int)(((long)N_EDGES * 128 + 255) / 256);

  // layer 0 (din=8, K=32)
  fill_zero_k<<<zeroBlocks, 256, 0, stream>>>(agg, aggN);
  scatter_k<<<scat8, 256, 0, stream>>>(srcp, dstp, hA, agg, 8, N_EDGES * 8);
  sage_wmma_k<8, 1, 1, 8, 128, 128><<<gemm_blocks, 256, 0, stream>>>(agg, deg, hA, W0, bs[0], hB);

  // layer 1 (din=128, K=256)
  fill_zero_k<<<zeroBlocks, 256, 0, stream>>>(agg, aggN);
  scatter_k<<<scat128, 256, 0, stream>>>(srcp, dstp, hB, agg, 128, N_EDGES * 128);
  sage_wmma_k<128, 8, 1, 8, 128, 128><<<gemm_blocks, 256, 0, stream>>>(agg, deg, hB, W1, bs[1], hA);

  // layer 2
  fill_zero_k<<<zeroBlocks, 256, 0, stream>>>(agg, aggN);
  scatter_k<<<scat128, 256, 0, stream>>>(srcp, dstp, hA, agg, 128, N_EDGES * 128);
  sage_wmma_k<128, 8, 1, 8, 128, 128><<<gemm_blocks, 256, 0, stream>>>(agg, deg, hA, W2, bs[2], hB);

  // layer 3 (dout=1, no relu) -> gout
  fill_zero_k<<<zeroBlocks, 256, 0, stream>>>(agg, aggN);
  scatter_k<<<scat128, 256, 0, stream>>>(srcp, dstp, hB, agg, 128, N_EDGES * 128);
  sage_wmma_k<128, 8, 0, 1, 1, 1><<<gemm_blocks, 256, 0, stream>>>(agg, deg, hB, W3, bs[3], gout);
}
